// TaskRecurrentEncoder_14757507629763
// MI455X (gfx1250) — compile-verified, matrix operand order checked by reading, not run on hardware
//
#include <hip/hip_runtime.h>

// ---------------------------------------------------------------------------
// Problem constants (from reference)
// ---------------------------------------------------------------------------
#define BS   2048
#define NE   64
#define NA   16
#define DIN  128
#define HDIM 512
#define ADIM 256
#define NH   8
#define EDIM 32            // ADIM / NH
#define RL   64            // HDIM / 8
#define OUT  128
#define MROWS   (BS * NE)  // 131072 embed rows
#define AROWS   (BS * NA)  // 32768 agent rows
#define XK      (HDIM + ADIM + RL)   // 832 GRU input width
#define G3      (3 * HDIM)           // 1536 gate width
#define KVW     (2 * ADIM)           // 512: fused K|V width

typedef __bf16 bf16;
typedef __attribute__((ext_vector_type(16))) __bf16 bf16x16;
typedef __attribute__((ext_vector_type(4)))  __bf16 bf16x4;
typedef __attribute__((ext_vector_type(8))) float  floatx8;

#define LDST 40   // padded LDS row stride (bf16): 80B rows, conflict-free b128 frag reads

// ---------------------------------------------------------------------------
// CDNA5 async global->LDS copy (ASYNCcnt path). One b128 per lane.
// Flat-address low 32 bits == wave-relative LDS byte offset (ISA 10.2).
// ---------------------------------------------------------------------------
static __device__ __forceinline__ void async_b128(void* lds_dst, const void* gsrc) {
  unsigned l = (unsigned)(unsigned long long)lds_dst;
  asm volatile("global_load_async_to_lds_b128 %0, %1, off"
               :: "v"(l), "v"(gsrc) : "memory");
}
static __device__ __forceinline__ void async_wait0() {
  asm volatile("s_wait_asynccnt 0" ::: "memory");
}

// Load one 16x32 bf16 WMMA fragment from padded LDS.
// CDNA5 16-bit A/B layout: lanes 0-15 -> K 0..7 & 16..23 ; lanes 16-31 -> K 8..15 & 24..31
static __device__ __forceinline__ bf16x16 ld_frag(const bf16* p) {
  bf16x16 a;
  uint4* d = reinterpret_cast<uint4*>(&a);
  d[0] = *reinterpret_cast<const uint4*>(p);
  d[1] = *reinterpret_cast<const uint4*>(p + 16);
  return a;
}

static __device__ __forceinline__ floatx8 wmma_bf16(bf16x16 a, bf16x16 b, floatx8 c) {
  return __builtin_amdgcn_wmma_f32_16x16x32_bf16(false, a, false, b, (short)0, c, false, false);
}

static __device__ __forceinline__ void cvt_store8(bf16* dst, float4 f0, float4 f1) {
  bf16x4 h0; h0[0] = (bf16)f0.x; h0[1] = (bf16)f0.y; h0[2] = (bf16)f0.z; h0[3] = (bf16)f0.w;
  bf16x4 h1; h1[0] = (bf16)f1.x; h1[1] = (bf16)f1.y; h1[2] = (bf16)f1.z; h1[3] = (bf16)f1.w;
  *reinterpret_cast<bf16x4*>(dst)     = h0;
  *reinterpret_cast<bf16x4*>(dst + 4) = h1;
}

// ---------------------------------------------------------------------------
// Prep: transpose f32 weight (K x N) -> bf16 (N x K)
// ---------------------------------------------------------------------------
__global__ void transpose_to_bf16(const float* __restrict__ W, bf16* __restrict__ Wt,
                                  int K, int N) {
  int i = blockIdx.x * 256 + threadIdx.x;
  if (i >= K * N) return;
  int k = i / N, n = i - k * N;
  Wt[(unsigned)n * K + k] = (bf16)W[(unsigned)k * N + n];
}

// ---------------------------------------------------------------------------
// embed GEMM: A f32 (M x K) * Bt bf16 -> relu(+bias) -> C bf16, fused agent
// gather. Double-buffered LDS pipeline; A-tile f32 loads are register-
// pipelined (issue before WMMAs, cvt+ds_store after).
// ---------------------------------------------------------------------------
__global__ __launch_bounds__(256) void gemm_embed(
    const float* __restrict__ A, const bf16* __restrict__ Bt,
    const float* __restrict__ bias, bf16* __restrict__ C, bf16* __restrict__ CA,
    int M, int N, int K) {
  __shared__ bf16 sA[2][128 * LDST];
  __shared__ bf16 sB[2][128 * LDST];
  const int t = threadIdx.x;
  const int wid = t >> 5, lane = t & 31, hf = lane >> 4, l = lane & 15;
  const int wm = wid & 1, wn = wid >> 1;
  const unsigned m0 = blockIdx.x * 128, n0 = blockIdx.y * 128;
  const int ra = t >> 3, ca = (t & 7) * 4;      // A-tile: 128 rows x 32, float4 slots
  const int rb = t >> 2, cb = (t & 3) * 8;      // B-tile: 8-elem chunks
  floatx8 acc[4][2];
  for (int i = 0; i < 4; ++i)
    for (int j = 0; j < 2; ++j)
      for (int v = 0; v < 8; ++v) acc[i][j][v] = 0.f;

  auto loadA = [&](int k0, float4* f) {
#pragma unroll
    for (int i = 0; i < 4; ++i)
      f[i] = *reinterpret_cast<const float4*>(A + (m0 + ra + i * 32) * (unsigned)K + k0 + ca);
  };
  auto storeA = [&](int p, const float4* f) {
#pragma unroll
    for (int i = 0; i < 4; ++i) {
      bf16x4 hv; hv[0] = (bf16)f[i].x; hv[1] = (bf16)f[i].y;
      hv[2] = (bf16)f[i].z; hv[3] = (bf16)f[i].w;
      *reinterpret_cast<bf16x4*>(sA[p] + (ra + i * 32) * LDST + ca) = hv;
    }
  };
  auto loadB = [&](int k0, int p) {
#pragma unroll
    for (int i = 0; i < 2; ++i)
      async_b128(sB[p] + (rb + i * 64) * LDST + cb,
                 Bt + (n0 + rb + i * 64) * (unsigned)K + k0 + cb);
  };

  { // prime tile 0
    float4 f[4];
    loadA(0, f);
    loadB(0, 0);
    storeA(0, f);
    async_wait0();
    __syncthreads();
  }
  int p = 0;
  const int klo = hf * 8;
  for (int k0 = 0; k0 < K; k0 += 32, p ^= 1) {
    const bool more = (k0 + 32) < K;
    float4 fn[4];
    if (more) { loadA(k0 + 32, fn); loadB(k0 + 32, p ^ 1); }
    bf16x16 afr[4], bfr[2];
#pragma unroll
    for (int i = 0; i < 4; ++i) afr[i] = ld_frag(sA[p] + (wm * 64 + i * 16 + l) * LDST + klo);
#pragma unroll
    for (int j = 0; j < 2; ++j) bfr[j] = ld_frag(sB[p] + (wn * 32 + j * 16 + l) * LDST + klo);
#pragma unroll
    for (int i = 0; i < 4; ++i)
#pragma unroll
      for (int j = 0; j < 2; ++j) acc[i][j] = wmma_bf16(afr[i], bfr[j], acc[i][j]);
    if (more) storeA(p ^ 1, fn);
    async_wait0();
    __syncthreads();
  }
#pragma unroll
  for (int j = 0; j < 2; ++j) {
    unsigned n = n0 + wn * 32 + j * 16 + l;
    float bv = bias[n];
#pragma unroll
    for (int i = 0; i < 4; ++i)
#pragma unroll
      for (int v = 0; v < 8; ++v) {
        unsigned m = m0 + wm * 64 + i * 16 + v + 8 * hf;
        float x = fmaxf(acc[i][j][v] + bv, 0.f);
        bf16 hx = (bf16)x;
        C[m * (unsigned)N + n] = hx;
        unsigned a = m & (NE - 1);
        if (a < NA) CA[((m >> 6) * NA + a) * (unsigned)HDIM + n] = hx;  // gather agents
      }
  }
}

// ---------------------------------------------------------------------------
// Shared bf16 GEMM main loop (double-buffered async pipeline).
// Fills acc[4][2] for a 128x128 tile at (m0, n0). Each caller uses it once.
// ---------------------------------------------------------------------------
static __device__ __forceinline__ void gemm16_main(
    const bf16* __restrict__ A, const bf16* __restrict__ Bt, int K,
    unsigned m0, unsigned n0, floatx8 (&acc)[4][2]) {
  __shared__ bf16 sA[2][128 * LDST];
  __shared__ bf16 sB[2][128 * LDST];
  const int t = threadIdx.x;
  const int wid = t >> 5, lane = t & 31, hf = lane >> 4, l = lane & 15;
  const int wm = wid & 1, wn = wid >> 1;
  const int rb = t >> 2, cb = (t & 3) * 8;
  for (int i = 0; i < 4; ++i)
    for (int j = 0; j < 2; ++j)
      for (int v = 0; v < 8; ++v) acc[i][j][v] = 0.f;

  auto loadAB = [&](int k0, int p) {
#pragma unroll
    for (int i = 0; i < 2; ++i) {
      async_b128(sA[p] + (rb + i * 64) * LDST + cb,
                 A + (m0 + rb + i * 64) * (unsigned)K + k0 + cb);
      async_b128(sB[p] + (rb + i * 64) * LDST + cb,
                 Bt + (n0 + rb + i * 64) * (unsigned)K + k0 + cb);
    }
  };
  loadAB(0, 0);
  async_wait0();
  __syncthreads();
  int p = 0;
  const int klo = hf * 8;
  for (int k0 = 0; k0 < K; k0 += 32, p ^= 1) {
    if (k0 + 32 < K) loadAB(k0 + 32, p ^ 1);
    bf16x16 afr[4], bfr[2];
#pragma unroll
    for (int i = 0; i < 4; ++i) afr[i] = ld_frag(sA[p] + (wm * 64 + i * 16 + l) * LDST + klo);
#pragma unroll
    for (int j = 0; j < 2; ++j) bfr[j] = ld_frag(sB[p] + (wn * 32 + j * 16 + l) * LDST + klo);
#pragma unroll
    for (int i = 0; i < 4; ++i)
#pragma unroll
      for (int j = 0; j < 2; ++j) acc[i][j] = wmma_bf16(afr[i], bfr[j], acc[i][j]);
    async_wait0();
    __syncthreads();
  }
}

// Q projection: +bias -> bf16
__global__ __launch_bounds__(256) void gemm_q(
    const bf16* __restrict__ A, const bf16* __restrict__ Bt,
    const float* __restrict__ bias, bf16* __restrict__ C, int M, int N, int K) {
  const int t = threadIdx.x;
  const int wid = t >> 5, lane = t & 31, hf = lane >> 4, l = lane & 15;
  const int wm = wid & 1, wn = wid >> 1;
  const unsigned m0 = blockIdx.x * 128, n0 = blockIdx.y * 128;
  floatx8 acc[4][2];
  gemm16_main(A, Bt, K, m0, n0, acc);
#pragma unroll
  for (int j = 0; j < 2; ++j) {
    unsigned n = n0 + wn * 32 + j * 16 + l;
    float bv = bias[n];
#pragma unroll
    for (int i = 0; i < 4; ++i)
#pragma unroll
      for (int v = 0; v < 8; ++v) {
        unsigned m = m0 + wm * 64 + i * 16 + v + 8 * hf;
        C[m * (unsigned)N + n] = (bf16)(acc[i][j][v] + bv);
      }
  }
}

// Fused K|V projection: Bt = [W_k^T ; W_v^T]; cols<256: +b_k, cols>=256: relu(+b_v)
__global__ __launch_bounds__(256) void gemm_kv(
    const bf16* __restrict__ A, const bf16* __restrict__ Bt,
    const float* __restrict__ biasK, const float* __restrict__ biasV,
    bf16* __restrict__ C, int M, int K) {
  const int t = threadIdx.x;
  const int wid = t >> 5, lane = t & 31, hf = lane >> 4, l = lane & 15;
  const int wm = wid & 1, wn = wid >> 1;
  const unsigned m0 = blockIdx.x * 128, n0 = blockIdx.y * 128;
  floatx8 acc[4][2];
  gemm16_main(A, Bt, K, m0, n0, acc);
#pragma unroll
  for (int j = 0; j < 2; ++j) {
    unsigned n = n0 + wn * 32 + j * 16 + l;
    bool isV = n >= ADIM;
    float bv = isV ? biasV[n - ADIM] : biasK[n];
#pragma unroll
    for (int i = 0; i < 4; ++i)
#pragma unroll
      for (int v = 0; v < 8; ++v) {
        unsigned m = m0 + wm * 64 + i * 16 + v + 8 * hf;
        float x = acc[i][j][v] + bv;
        if (isV) x = fmaxf(x, 0.f);
        C[m * (unsigned)KVW + n] = (bf16)x;
      }
  }
}

// params GEMM: +bias, scenario row-mask -> f32
__global__ __launch_bounds__(256) void gemm_params(
    const bf16* __restrict__ A, const bf16* __restrict__ Bt,
    const float* __restrict__ bias, const unsigned char* __restrict__ scen,
    float* __restrict__ C, int M, int N, int K) {
  const int t = threadIdx.x;
  const int wid = t >> 5, lane = t & 31, hf = lane >> 4, l = lane & 15;
  const int wm = wid & 1, wn = wid >> 1;
  const unsigned m0 = blockIdx.x * 128, n0 = blockIdx.y * 128;
  floatx8 acc[4][2];
  gemm16_main(A, Bt, K, m0, n0, acc);
#pragma unroll
  for (int j = 0; j < 2; ++j) {
    unsigned n = n0 + wn * 32 + j * 16 + l;
    float bv = bias[n];
#pragma unroll
    for (int i = 0; i < 4; ++i)
#pragma unroll
      for (int v = 0; v < 8; ++v) {
        unsigned m = m0 + wm * 64 + i * 16 + v + 8 * hf;
        float x = acc[i][j][v] + bv;
        if (scen[((m >> 4) << 6) + (m & 15)]) x = 0.f;  // scenario_mask[b, a]
        C[m * (unsigned)N + n] = x;
      }
  }
}

// ---------------------------------------------------------------------------
// rew = relu(pre_rew * W_rew + b_rew)   (BS x RL, f32)
// ---------------------------------------------------------------------------
__global__ void rew_kernel(const float* __restrict__ prew, const float* __restrict__ Wr,
                           const float* __restrict__ br, float* __restrict__ rew) {
  int i = blockIdx.x * 256 + threadIdx.x;
  int b = i >> 6, j = i & 63;
  rew[i] = fmaxf(prew[b] * Wr[j] + br[j], 0.f);
}

// ---------------------------------------------------------------------------
// Attention: one block per batch; fused KV tile async-staged to LDS (64KB),
// masked softmax in registers, thread = (head, query). ~1% of FLOPs -> VALU.
// ---------------------------------------------------------------------------
__global__ __launch_bounds__(128) void attention_kernel(
    const bf16* __restrict__ Qb, const bf16* __restrict__ KV,
    const unsigned char* __restrict__ obs, bf16* __restrict__ atten) {
  __shared__ bf16 sKV[NE * KVW];   // 64 rows x (K|V 512) = 64KB
  const unsigned b = blockIdx.x;
  const int t = threadIdx.x;
  const bf16* src = KV + b * (unsigned)(NE * KVW);
#pragma unroll
  for (int i = 0; i < (NE * KVW) / (8 * 128); ++i) {
    int s = t + i * 128;
    async_b128(sKV + s * 8, src + s * 8);
  }
  async_wait0();
  __syncthreads();
  const int h = t >> 4, q = t & 15;
  float qv[EDIM];
  const bf16* qp = Qb + (b * NA + q) * (unsigned)ADIM + h * EDIM;
#pragma unroll
  for (int e = 0; e < EDIM; ++e) qv[e] = (float)qp[e];
  const float scale = 0.17677669529663687f;  // 1/sqrt(32)
  float sc[NE];
  float mx = -INFINITY;
  const unsigned char* ob = obs + (b * NE + q) * (unsigned)NE;
  for (int k = 0; k < NE; ++k) {
    float s;
    if ((k == q) || ob[k]) {
      s = -INFINITY;
    } else {
      float d = 0.f;
      const bf16* kp = sKV + k * KVW + h * EDIM;
#pragma unroll
      for (int e = 0; e < EDIM; ++e) d += qv[e] * (float)kp[e];
      s = d * scale;
    }
    sc[k] = s;
    mx = fmaxf(mx, s);
  }
  float acc[EDIM];
#pragma unroll
  for (int e = 0; e < EDIM; ++e) acc[e] = 0.f;
  if (mx > -INFINITY) {
    float sum = 0.f;
    for (int k = 0; k < NE; ++k) { float w = __expf(sc[k] - mx); sc[k] = w; sum += w; }
    float inv = 1.f / sum;
    for (int k = 0; k < NE; ++k) {
      float w = sc[k] * inv;
      const bf16* vp = sKV + k * KVW + ADIM + h * EDIM;
#pragma unroll
      for (int e = 0; e < EDIM; ++e) acc[e] += w * (float)vp[e];
    }
  }
  bf16* op = atten + (b * NA + q) * (unsigned)ADIM + h * EDIM;
#pragma unroll
  for (int e = 0; e < EDIM; ++e) op[e] = (bf16)acc[e];
}

// ---------------------------------------------------------------------------
// Fused GRU: per wave, one 16-col tile of H with all 3 gates for gi AND gh
// (12 f32 accumulators). x = [embedA | atten | rew] assembled on the fly.
// Double-buffered async pipeline; pre_hidden converts register-pipelined.
// block: 256 thr = 8 waves = 2(m) x 4(n); tile 64 rows x 64 H-cols.
// ---------------------------------------------------------------------------
__global__ __launch_bounds__(256) void gru_kernel(
    const bf16* __restrict__ embedA, const bf16* __restrict__ atten,
    const float* __restrict__ rew, const float* __restrict__ preh,
    const bf16* __restrict__ WtIH, const float* __restrict__ bih,
    const bf16* __restrict__ WtHH, const float* __restrict__ bhh,
    const unsigned char* __restrict__ scen,
    float* __restrict__ hout, bf16* __restrict__ hbuf) {
  __shared__ bf16 sX[2][64 * LDST];
  __shared__ bf16 sG[2][3][64 * LDST];
  const int t = threadIdx.x;
  const int wid = t >> 5, lane = t & 31, hf = lane >> 4, l = lane & 15;
  const int wm = wid & 1, wn = wid >> 1;
  const unsigned r0 = blockIdx.x * 64;    // rows over BS*NA
  const unsigned c0 = blockIdx.y * 64;    // cols over HDIM
  floatx8 acci[2][3], acch[2][3];
  for (int i = 0; i < 2; ++i)
    for (int g = 0; g < 3; ++g)
      for (int v = 0; v < 8; ++v) { acci[i][g][v] = 0.f; acch[i][g][v] = 0.f; }
  const int lr = t >> 2;            // 0..63 tile row
  const int lc = (t & 3) * 8;       // col chunk (8 elems)
  const unsigned gr = r0 + lr;
  const int klo = hf * 8;

  auto load_g = [&](const bf16* Wt, int ldw, int k0, int p) {
#pragma unroll
    for (int g = 0; g < 3; ++g)
      async_b128(sG[p][g] + lr * LDST + lc,
                 Wt + (unsigned)(g * HDIM + c0 + lr) * ldw + k0 + lc);
  };
  // phase-1 x tile: embedA | atten async; rew chunk converted inline (2/26 iters)
  auto load_x1 = [&](int k0, int p) {
    int k = k0 + lc;
    if (k0 < HDIM) {
      async_b128(sX[p] + lr * LDST + lc, embedA + gr * (unsigned)HDIM + k);
    } else if (k0 < HDIM + ADIM) {
      async_b128(sX[p] + lr * LDST + lc, atten + gr * (unsigned)ADIM + (k - HDIM));
    } else {
      const float* rp = rew + (gr >> 4) * (unsigned)RL + (k - HDIM - ADIM);
      float4 f0 = *reinterpret_cast<const float4*>(rp);
      float4 f1 = *reinterpret_cast<const float4*>(rp + 4);
      cvt_store8(sX[p] + lr * LDST + lc, f0, f1);
    }
  };

  // ---- phase 1: gi = x @ W_ih   (K = 832) ----
  load_x1(0, 0);
  load_g(WtIH, XK, 0, 0);
  async_wait0();
  __syncthreads();
  int p = 0;
  for (int k0 = 0; k0 < XK; k0 += 32, p ^= 1) {
    if (k0 + 32 < XK) { load_x1(k0 + 32, p ^ 1); load_g(WtIH, XK, k0 + 32, p ^ 1); }
    bf16x16 afr[2], bfr[3];
#pragma unroll
    for (int mi = 0; mi < 2; ++mi)
      afr[mi] = ld_frag(sX[p] + (wm * 32 + mi * 16 + l) * LDST + klo);
#pragma unroll
    for (int g = 0; g < 3; ++g) bfr[g] = ld_frag(sG[p][g] + (wn * 16 + l) * LDST + klo);
#pragma unroll
    for (int mi = 0; mi < 2; ++mi)
#pragma unroll
      for (int g = 0; g < 3; ++g) acci[mi][g] = wmma_bf16(afr[mi], bfr[g], acci[mi][g]);
    async_wait0();
    __syncthreads();
  }

  // ---- phase 2: gh = pre_hidden @ W_hh   (K = 512) ----
  {
    const float* pp = preh + gr * (unsigned)HDIM + lc;
    float4 f0 = *reinterpret_cast<const float4*>(pp);
    float4 f1 = *reinterpret_cast<const float4*>(pp + 4);
    cvt_store8(sX[0] + lr * LDST + lc, f0, f1);
    load_g(WtHH, HDIM, 0, 0);
    async_wait0();
    __syncthreads();
  }
  p = 0;
  for (int k0 = 0; k0 < HDIM; k0 += 32, p ^= 1) {
    const bool more = (k0 + 32) < HDIM;
    float4 f0n, f1n;
    if (more) {
      const float* pp = preh + gr * (unsigned)HDIM + k0 + 32 + lc;
      f0n = *reinterpret_cast<const float4*>(pp);
      f1n = *reinterpret_cast<const float4*>(pp + 4);
      load_g(WtHH, HDIM, k0 + 32, p ^ 1);
    }
    bf16x16 afr[2], bfr[3];
#pragma unroll
    for (int mi = 0; mi < 2; ++mi)
      afr[mi] = ld_frag(sX[p] + (wm * 32 + mi * 16 + l) * LDST + klo);
#pragma unroll
    for (int g = 0; g < 3; ++g) bfr[g] = ld_frag(sG[p][g] + (wn * 16 + l) * LDST + klo);
#pragma unroll
    for (int mi = 0; mi < 2; ++mi)
#pragma unroll
      for (int g = 0; g < 3; ++g) acch[mi][g] = wmma_bf16(afr[mi], bfr[g], acch[mi][g]);
    if (more) cvt_store8(sX[p ^ 1] + lr * LDST + lc, f0n, f1n);
    async_wait0();
    __syncthreads();
  }

  // ---- epilogue: gate math, scenario mask, dual store ----
  const unsigned n = c0 + wn * 16 + l;
  const float bir = bih[n], biz = bih[n + HDIM], bin = bih[n + 2 * HDIM];
  const float bhr = bhh[n], bhz = bhh[n + HDIM], bhn = bhh[n + 2 * HDIM];
#pragma unroll
  for (int mi = 0; mi < 2; ++mi)
#pragma unroll
    for (int v = 0; v < 8; ++v) {
      unsigned row = r0 + wm * 32 + mi * 16 + v + 8 * hf;
      float sr = (acci[mi][0][v] + bir) + (acch[mi][0][v] + bhr);
      float sz = (acci[mi][1][v] + biz) + (acch[mi][1][v] + bhz);
      float rg = 1.f / (1.f + __expf(-sr));
      float zg = 1.f / (1.f + __expf(-sz));
      float ng = tanhf((acci[mi][2][v] + bin) + rg * (acch[mi][2][v] + bhn));
      float hp = preh[row * (unsigned)HDIM + n];
      float hv = (1.f - zg) * ng + zg * hp;
      if (scen[((row >> 4) << 6) + (row & 15)]) hv = 0.f;
      hout[row * (unsigned)HDIM + n] = hv;
      hbuf[row * (unsigned)HDIM + n] = (bf16)hv;
    }
}

// ---------------------------------------------------------------------------
// Host side
// ---------------------------------------------------------------------------
extern "C" void kernel_launch(void* const* d_in, const int* in_sizes, int n_in,
                              void* d_out, int out_size, void* d_ws, size_t ws_size,
                              hipStream_t stream) {
  const float* state = (const float*)d_in[0];
  const float* prew  = (const float*)d_in[1];
  const unsigned char* obs  = (const unsigned char*)d_in[2];
  const unsigned char* scen = (const unsigned char*)d_in[3];
  const float* preh  = (const float*)d_in[4];
  // d_in[5] = task_index (unused by reference math)
  const float* W_state = (const float*)d_in[6];
  const float* b_state = (const float*)d_in[7];
  const float* W_rew   = (const float*)d_in[8];
  const float* b_rew   = (const float*)d_in[9];
  const float* W_q     = (const float*)d_in[10];
  const float* b_q     = (const float*)d_in[11];
  const float* W_k     = (const float*)d_in[12];
  const float* b_k     = (const float*)d_in[13];
  const float* W_v     = (const float*)d_in[14];
  const float* b_v     = (const float*)d_in[15];
  const float* W_ih    = (const float*)d_in[16];
  const float* b_ih    = (const float*)d_in[17];
  const float* W_hh    = (const float*)d_in[18];
  const float* b_hh    = (const float*)d_in[19];
  const float* W_out   = (const float*)d_in[20];
  const float* b_out   = (const float*)d_in[21];

  float* params = (float*)d_out;                       // (BS*NA, OUT)
  float* hout   = params + (size_t)AROWS * OUT;        // (BS*NA, HDIM)

  // workspace carve-out (256B aligned)
  char* wsp = (char*)d_ws;
  auto carve = [&](size_t bytes) -> char* {
    char* p = wsp;
    wsp += (bytes + 255) & ~(size_t)255;
    return p;
  };
  bf16* WtS  = (bf16*)carve((size_t)HDIM * DIN * 2);
  bf16* WtKV = (bf16*)carve((size_t)KVW * HDIM * 2);   // [W_k^T ; W_v^T]
  bf16* WtQ  = (bf16*)carve((size_t)ADIM * HDIM * 2);
  bf16* WtIH = (bf16*)carve((size_t)G3 * XK * 2);
  bf16* WtHH = (bf16*)carve((size_t)G3 * HDIM * 2);
  bf16* WtO  = (bf16*)carve((size_t)OUT * HDIM * 2);
  bf16* embed  = (bf16*)carve((size_t)MROWS * HDIM * 2);
  bf16* embedA = (bf16*)carve((size_t)AROWS * HDIM * 2);
  bf16* KVbuf  = (bf16*)carve((size_t)MROWS * KVW * 2);
  bf16* Qbuf   = (bf16*)carve((size_t)AROWS * ADIM * 2);
  bf16* atten  = (bf16*)carve((size_t)AROWS * ADIM * 2);
  float* rew   = (float*)carve((size_t)BS * RL * 4);
  bf16* hbuf   = (bf16*)carve((size_t)AROWS * HDIM * 2);

  // 1) weight prep (tiny)
  auto tr = [&](const float* W, bf16* Wt, int K, int N) {
    transpose_to_bf16<<<(K * N + 255) / 256, 256, 0, stream>>>(W, Wt, K, N);
  };
  tr(W_state, WtS, DIN, HDIM);
  tr(W_k, WtKV, HDIM, ADIM);                       // rows [0,256)
  tr(W_v, WtKV + (size_t)ADIM * HDIM, HDIM, ADIM); // rows [256,512)
  tr(W_q, WtQ, HDIM, ADIM);
  tr(W_ih, WtIH, XK, G3);
  tr(W_hh, WtHH, HDIM, G3);
  tr(W_out, WtO, HDIM, OUT);

  // 2) embed = relu(state @ W_state + b), fused agent gather
  gemm_embed<<<dim3(MROWS / 128, HDIM / 128), 256, 0, stream>>>(
      state, WtS, b_state, embed, embedA, MROWS, HDIM, DIN);

  // 3) fused K|V projection (embed read once)
  gemm_kv<<<dim3(MROWS / 128, KVW / 128), 256, 0, stream>>>(
      embed, WtKV, b_k, b_v, KVbuf, MROWS, HDIM);

  // 4) Q projection
  gemm_q<<<dim3(AROWS / 128, ADIM / 128), 256, 0, stream>>>(
      embedA, WtQ, b_q, Qbuf, AROWS, ADIM, HDIM);

  // 5) reward MLP (tiny)
  rew_kernel<<<(BS * RL) / 256, 256, 0, stream>>>(prew, W_rew, b_rew, rew);

  // 6) masked attention
  attention_kernel<<<BS, 128, 0, stream>>>(Qbuf, KVbuf, obs, atten);

  // 7) fused GRU -> h (f32 out) + h (bf16 scratch)
  gru_kernel<<<dim3(AROWS / 64, HDIM / 64), 256, 0, stream>>>(
      embedA, atten, rew, preh, WtIH, b_ih, WtHH, b_hh, scen, hout, hbuf);

  // 8) params = h @ W_out + b_out, scenario-masked (f32 out)
  gemm_params<<<dim3(AROWS / 128, OUT / 128), 256, 0, stream>>>(
      hbuf, WtO, b_out, scen, params, AROWS, OUT, HDIM);

  (void)in_sizes; (void)n_in; (void)out_size; (void)ws_size;
}
